// binary_classification_model_16544214024209
// MI455X (gfx1250) — matrix operand order repaired and checked by reading, not compiled, not run on hardware
//
#include <hip/hip_runtime.h>

// ---------------------------------------------------------------------------
// MI455X / gfx1250: all GEMM stages on v_wmma_f32_16x16x32_bf16, operands
// pre-converted to bf16 in workspace with WMMA-friendly layouts so every
// fragment is two 16-byte vector loads. The conv stage (180 GFLOP, dominant)
// stages its shared Wconv A-tile into LDS once per block with
// GLOBAL_LOAD_ASYNC_TO_LDS_B128 (ASYNCcnt / s_wait_asynccnt path).
// ---------------------------------------------------------------------------

typedef unsigned short u16;
typedef __bf16 bf16_t;
typedef bf16_t v16bf  __attribute__((ext_vector_type(16)));
typedef float  v8f    __attribute__((ext_vector_type(8)));
typedef u16    u16x4  __attribute__((ext_vector_type(4)));
typedef u16    u16x8  __attribute__((ext_vector_type(8)));
typedef u16    u16x16 __attribute__((ext_vector_type(16)));
typedef float  f32x4  __attribute__((ext_vector_type(4)));

#define B_  4
#define S_  2048
#define E_  1024
#define P_  512
#define KW_ 7
#define NK_ 6
#define H_  8
#define DL_ 64

__device__ __forceinline__ u16 f2bfbits(float f) {
  union { float f; unsigned u; } v; v.f = f;
  unsigned r = v.u + 0x7FFFu + ((v.u >> 16) & 1u);   // round-to-nearest-even
  return (u16)(r >> 16);
}

__device__ __forceinline__ v8f wmma_bf16(v16bf a, v16bf b, v8f c) {
  return __builtin_amdgcn_wmma_f32_16x16x32_bf16(false, a, false, b, (short)0, c,
                                                 false, false);
}

// A-fragment (16x32 bf16, ISA 7.12.2): lane(hi) holds K runs {hi*8..+8} and
// {16+hi*8..+8} of its row -> two contiguous 16B loads.
__device__ __forceinline__ v16bf fragA(const u16* rowbase, int hi) {
  u16x8 r0 = *(const u16x8*)(rowbase + hi * 8);
  u16x8 r1 = *(const u16x8*)(rowbase + 16 + hi * 8);
  u16x16 c = __builtin_shufflevector(r0, r1, 0, 1, 2, 3, 4, 5, 6, 7,
                                     8, 9, 10, 11, 12, 13, 14, 15);
  return __builtin_bit_cast(v16bf, c);
}
// B-fragment (32x16 bf16): lane(hi) holds K runs {hi*8..+8} and {16+hi*8..+8}
// of its column -> two 16B loads from a K-contiguous (transposed) row.
__device__ __forceinline__ v16bf fragB(const u16* rowbase, int hi) {
  u16x8 r0 = *(const u16x8*)(rowbase + hi * 8);
  u16x8 r1 = *(const u16x8*)(rowbase + 16 + hi * 8);
  u16x16 c = __builtin_shufflevector(r0, r1, 0, 1, 2, 3, 4, 5, 6, 7,
                                     8, 9, 10, 11, 12, 13, 14, 15);
  return __builtin_bit_cast(v16bf, c);
}
__device__ __forceinline__ v16bf zero_frag() {
  u16x16 z = {};
  return __builtin_bit_cast(v16bf, z);
}

// ---- CDNA5 async global->LDS copy (ISA §15.18.3, ASYNCcnt) ----------------
// VDST VGPR carries the wave-relative LDS byte offset; generic shared pointer
// low 32 bits are exactly that offset (flat aperture mapping, ISA §10.2).
__device__ __forceinline__ void async_g2l_b128(u16* lds, const u16* gsrc) {
  unsigned loff = (unsigned)(unsigned long long)lds;
  asm volatile("global_load_async_to_lds_b128 %0, %1, off"
               :: "v"(loff), "v"(gsrc)
               : "memory");
}
__device__ __forceinline__ void wait_async0() {
  asm volatile("s_wait_asynccnt 0" ::: "memory");
}

// ---------------------------------------------------------------------------
// Conversion passes (one time, f32 -> bf16)
// ---------------------------------------------------------------------------
__global__ void cvt4_kernel(const float* __restrict__ in, u16* __restrict__ out,
                            int n4) {
  int i = blockIdx.x * blockDim.x + threadIdx.x;
  if (i >= n4) return;
  f32x4 v = *(const f32x4*)(in + (size_t)i * 4);
  u16x4 o;
  #pragma unroll
  for (int j = 0; j < 4; ++j) o[j] = f2bfbits(v[j]);
  *(u16x4*)(out + (size_t)i * 4) = o;
}

// Wconv [k][po][pi][t] f32  ->  [k][t][po][pi] bf16 (K-contiguous A operand)
__global__ void cvt_wconv_kernel(const float* __restrict__ w,
                                 u16* __restrict__ out) {
  int i = blockIdx.x * blockDim.x + threadIdx.x;
  const int N = NK_ * KW_ * P_ * P_;
  if (i >= N) return;
  int pi = i & (P_ - 1);
  int po = (i >> 9) & (P_ - 1);
  int tt = (i / (P_ * P_)) % KW_;
  int k  = i / (P_ * P_ * KW_);
  out[i] = f2bfbits(w[(((size_t)k * P_ + po) * P_ + pi) * KW_ + tt]);
}

__global__ void zero_rep_kernel(float* rep) { rep[threadIdx.x] = 0.0f; }

// ---------------------------------------------------------------------------
// Stage 1: proj = Wp @ x[b]^T + bp, masked; stored TRANSPOSED bf16 [b][s][p]
// One wave -> 32x32 tile (2x2 WMMA register blocking), K-loop over E.
// ---------------------------------------------------------------------------
__global__ void proj_kernel(const u16*   __restrict__ xb,
                            const int*   __restrict__ seqlen,
                            const u16*   __restrict__ wpb,
                            const float* __restrict__ bp,
                            u16*         __restrict__ projt) {
  int wave = (blockIdx.x * blockDim.x + threadIdx.x) >> 5;
  int lane = threadIdx.x & 31;
  int st = wave % (S_ / 32);
  int pt = (wave / (S_ / 32)) % (P_ / 32);
  int b  = wave / ((S_ / 32) * (P_ / 32));
  int p0 = pt * 32, s0 = st * 32;
  int r = lane & 15, hi = lane >> 4;

  v8f acc[2][2] = {};
  #pragma unroll 1
  for (int k0 = 0; k0 < E_; k0 += 32) {
    v16bf a0 = fragA(wpb + (size_t)(p0 + r) * E_ + k0, hi);
    v16bf a1 = fragA(wpb + (size_t)(p0 + 16 + r) * E_ + k0, hi);
    v16bf b0 = fragB(xb + ((size_t)b * S_ + s0 + r) * E_ + k0, hi);
    v16bf b1 = fragB(xb + ((size_t)b * S_ + s0 + 16 + r) * E_ + k0, hi);
    acc[0][0] = wmma_bf16(a0, b0, acc[0][0]);
    acc[0][1] = wmma_bf16(a0, b1, acc[0][1]);
    acc[1][0] = wmma_bf16(a1, b0, acc[1][0]);
    acc[1][1] = wmma_bf16(a1, b1, acc[1][1]);
  }

  int L = seqlen[b];
  #pragma unroll
  for (int pm = 0; pm < 2; ++pm)
    #pragma unroll
    for (int sn = 0; sn < 2; ++sn) {
      int s = s0 + sn * 16 + r;
      float mask = ((s >= 1) && (s <= L)) ? 1.0f : 0.0f;   // branchless epilogue
      u16x8 o;
      #pragma unroll
      for (int v = 0; v < 8; ++v) {
        int p = p0 + pm * 16 + hi * 8 + v;
        o[v] = f2bfbits((acc[pm][sn][v] + bp[p]) * mask);
      }
      // 8 consecutive p for this lane -> one 16B store
      *(u16x8*)(projt + ((size_t)b * S_ + s) * P_ + p0 + pm * 16 + hi * 8) = o;
    }
}

// ---------------------------------------------------------------------------
// Stage 2: conv = mean_k relu(conv1d_k(proj)+bconv_k); 6 x 7 x (P/32) K=32
// WMMAs per 32x32 tile (2x2 blocking). All 8 waves of a block share (b, p0),
// so the Wconv A-tile (32 rows x 512 pi = 32KB bf16) is staged ONCE per block
// into LDS with async global->LDS copies, then read back with ds_read_b128.
// Outputs stored rearranged bf16:
//   convb[b][h][s][l]  (Q / A operand),  convt[b][h][l][t]  (V / B operand)
// ---------------------------------------------------------------------------
__global__ void conv_kernel(const u16*   __restrict__ projt,
                            const u16*   __restrict__ wcb,
                            const float* __restrict__ bconv,
                            u16*         __restrict__ convb,
                            u16*         __restrict__ convt) {
  __shared__ u16 Atile[32 * P_];                    // 32 KB
  int tid  = threadIdx.x;
  int wave = tid >> 5;
  int lane = tid & 31;
  // 8 consecutive s-tiles per block share (b, pt)
  int blk    = blockIdx.x;
  int stBase = (blk % ((S_ / 32) / 8)) * 8;
  int pt     = (blk / ((S_ / 32) / 8)) % (P_ / 32);
  int b      = blk / (((S_ / 32) / 8) * (P_ / 32));
  int st = stBase + wave;
  int p0 = pt * 32, s0 = st * 32;
  int r = lane & 15, hi = lane >> 4;

  v8f tot[2][2] = {};
  #pragma unroll 1
  for (int k = 0; k < NK_; ++k) {
    v8f acc[2][2] = {};
    #pragma unroll 1
    for (int t = 0; t < KW_; ++t) {
      // ---- stage A-tile: Wconv[k][t][p0..p0+31][0..511] -> LDS (async) ----
      __syncthreads();                              // previous tile fully read
      {
        const u16* src = wcb + (size_t)(k * KW_ + t) * P_ * P_ + (size_t)p0 * P_;
        #pragma unroll
        for (int c = 0; c < 8; ++c) {               // 2048 x 16B chunks / 256 thr
          int idx = (c * 256 + tid) * 8;            // u16 offset, 16B aligned
          async_g2l_b128(&Atile[idx], src + idx);
        }
        wait_async0();                              // drain ASYNCcnt
      }
      __syncthreads();                              // tile visible to all waves

      int sA = s0 + r + t - (KW_ / 2);
      int sB = sA + 16;
      #pragma unroll 1
      for (int pi0 = 0; pi0 < P_; pi0 += 32) {
        v16bf a0 = fragA(&Atile[(size_t)r * P_ + pi0], hi);
        v16bf a1 = fragA(&Atile[(size_t)(16 + r) * P_ + pi0], hi);
        v16bf b0 = (sA >= 0 && sA < S_)
                       ? fragB(projt + ((size_t)b * S_ + sA) * P_ + pi0, hi)
                       : zero_frag();
        v16bf b1 = (sB >= 0 && sB < S_)
                       ? fragB(projt + ((size_t)b * S_ + sB) * P_ + pi0, hi)
                       : zero_frag();
        acc[0][0] = wmma_bf16(a0, b0, acc[0][0]);
        acc[0][1] = wmma_bf16(a0, b1, acc[0][1]);
        acc[1][0] = wmma_bf16(a1, b0, acc[1][0]);
        acc[1][1] = wmma_bf16(a1, b1, acc[1][1]);
      }
    }
    #pragma unroll
    for (int pm = 0; pm < 2; ++pm)
      #pragma unroll
      for (int sn = 0; sn < 2; ++sn)
        #pragma unroll
        for (int v = 0; v < 8; ++v) {
          int po = p0 + pm * 16 + hi * 8 + v;
          float c = acc[pm][sn][v] + bconv[k * P_ + po];
          tot[pm][sn][v] += (c > 0.0f) ? c : 0.0f;  // relu per branch
        }
  }

  #pragma unroll
  for (int pm = 0; pm < 2; ++pm)
    #pragma unroll
    for (int sn = 0; sn < 2; ++sn) {
      int s = s0 + sn * 16 + r;
      #pragma unroll
      for (int v = 0; v < 8; ++v) {
        int po = p0 + pm * 16 + hi * 8 + v;
        int h = po & (H_ - 1);                      // 'b (l h) s -> b h s l'
        int l = po >> 3;
        u16 val = f2bfbits(tot[pm][sn][v] * (1.0f / (float)NK_));
        convb[(((size_t)b * H_ + h) * S_ + s) * DL_ + l] = val;
        convt[(((size_t)b * H_ + h) * DL_ + l) * S_ + s] = val;
      }
    }
}

// ---------------------------------------------------------------------------
// Stage 3: attention per (b,h,16-row s-block), one wave per workgroup.
// score(16x2048) via WMMA into 128KB LDS (320KB/WGP allows it), masked
// softmax per row (bf16 written over consumed f32 of the same row), then
// rep = attn x V via WMMA (K=2048), head-mean + valid(s) folded into
// a wave32 shfl_xor + atomic column reduction.
// ---------------------------------------------------------------------------
__global__ void attn_kernel(const u16*   __restrict__ convb,
                            const u16*   __restrict__ convt,
                            const u16*   __restrict__ wab,
                            const float* __restrict__ ba,
                            const int*   __restrict__ seqlen,
                            float*       __restrict__ rep_acc) {
  extern __shared__ float smem[];                   // 16 * 2048 f32 = 128 KB
  int lane = threadIdx.x & 31;
  int st = blockIdx.x % (S_ / 16);
  int h  = (blockIdx.x / (S_ / 16)) % H_;
  int b  = blockIdx.x / ((S_ / 16) * H_);
  int s0 = st * 16;
  int r = lane & 15, hi = lane >> 4;
  int L = seqlen[b];
  const u16* Q  = convb + ((size_t)b * H_ + h) * S_ * DL_;  // [s][l]
  const u16* Vt = convt + ((size_t)b * H_ + h) * DL_ * S_;  // [l][t]

  // Q fragments: rows s0..s0+15, K=l in two 32-chunks (contiguous l)
  v16bf aq[2];
  #pragma unroll
  for (int c = 0; c < 2; ++c)
    aq[c] = fragA(Q + (size_t)(s0 + r) * DL_ + c * 32, hi);

  // ---- score GEMM: score[s,t] = q . Wa[t,:] + ba[t], masked ----
  #pragma unroll 1
  for (int t0 = 0; t0 < S_; t0 += 16) {
    v8f sc = {};
    #pragma unroll
    for (int c = 0; c < 2; ++c) {
      v16bf bw = fragB(wab + (size_t)(t0 + r) * DL_ + c * 32, hi);
      sc = wmma_bf16(aq[c], bw, sc);
    }
    int t = t0 + r;
    bool tv = (t >= 1) && (t <= L);
    #pragma unroll
    for (int v = 0; v < 8; ++v)
      smem[(v + hi * 8) * S_ + t] = tv ? (sc[v] + ba[t]) : -1e9f;
  }
  __syncthreads();

  // ---- masked softmax per row; bf16 attn written over this row's own f32 ----
  if (lane < 16) {
    float* row = smem + lane * S_;
    u16*   arow = (u16*)row;               // bf16 idx t -> f32 slot t/2 (consumed)
    float m = -3.4e38f;
    for (int t = 0; t < S_; ++t) m = fmaxf(m, row[t]);
    float sum = 0.0f;
    for (int t = 0; t < S_; ++t) sum += __expf(row[t] - m);
    float inv = 1.0f / sum;
    for (int t = 0; t < S_; ++t) arow[t] = f2bfbits(__expf(row[t] - m) * inv);
  }
  __syncthreads();

  // ---- rep GEMM: rep[16s x 64l] = attn(16x2048) x V(2048x64) ----
  v8f racc[4] = {};
  #pragma unroll 1
  for (int kc = 0; kc < S_ / 32; ++kc) {
    v16bf aat = fragA((const u16*)(smem + r * S_) + kc * 32, hi);
    #pragma unroll
    for (int nt = 0; nt < 4; ++nt) {
      v16bf bv = fragB(Vt + (size_t)(nt * 16 + r) * S_ + kc * 32, hi);
      racc[nt] = wmma_bf16(aat, bv, racc[nt]);
    }
  }

  // mean over heads (1/H) + valid(s) mask + sum over s -> rep_acc[b, l]
  #pragma unroll
  for (int nt = 0; nt < 4; ++nt) {
    float partial = 0.0f;
    #pragma unroll
    for (int v = 0; v < 8; ++v) {
      int s = s0 + v + hi * 8;
      float w = ((s >= 1) && (s <= L)) ? (1.0f / (float)H_) : 0.0f;
      partial += racc[nt][v] * w;
    }
    partial += __shfl_xor(partial, 16, 32);         // combine half-waves (wave32)
    if (hi == 0) atomicAdd(&rep_acc[b * DL_ + nt * 16 + r], partial);
  }
}

// ---------------------------------------------------------------------------
// Stage 4: tiny MLP head + softmax + argmax (one wave)
// ---------------------------------------------------------------------------
__global__ void head_kernel(const float* __restrict__ rep,
                            const float* __restrict__ Wc1,
                            const float* __restrict__ bc1,
                            const float* __restrict__ Wc2,
                            const float* __restrict__ bc2,
                            float*       __restrict__ out) {
  int lane = threadIdx.x;
  for (int b = 0; b < B_; ++b) {
    float h1 = 0.0f;                                // lane j computes h1[j]
    for (int i = 0; i < DL_; ++i) h1 += rep[b * DL_ + i] * Wc1[lane * DL_ + i];
    h1 += bc1[lane];
    h1 = (h1 > 0.0f) ? h1 : 0.01f * h1;             // leaky_relu(0.01)
    float p0 = h1 * Wc2[lane];
    float p1 = h1 * Wc2[32 + lane];
    for (int off = 16; off > 0; off >>= 1) {
      p0 += __shfl_down(p0, off, 32);
      p1 += __shfl_down(p1, off, 32);
    }
    if (lane == 0) {
      float l0 = p0 + bc2[0], l1 = p1 + bc2[1];
      float m = fmaxf(l0, l1);
      float e0 = __expf(l0 - m), e1 = __expf(l1 - m);
      float s = e0 + e1;
      out[b * 2 + 0] = e0 / s;
      out[b * 2 + 1] = e1 / s;
      out[2 * B_ + b] = (l1 > l0) ? 1.0f : 0.0f;    // argmax (ties -> 0)
    }
  }
}

// ---------------------------------------------------------------------------
extern "C" void kernel_launch(void* const* d_in, const int* in_sizes, int n_in,
                              void* d_out, int out_size, void* d_ws, size_t ws_size,
                              hipStream_t stream) {
  const float* x      = (const float*)d_in[0];
  const int*   seqlen = (const int*)  d_in[1];
  const float* Wp     = (const float*)d_in[2];
  const float* bp     = (const float*)d_in[3];
  const float* Wconv  = (const float*)d_in[4];
  const float* bconv  = (const float*)d_in[5];
  const float* Wa     = (const float*)d_in[6];
  const float* ba     = (const float*)d_in[7];
  const float* Wc1    = (const float*)d_in[8];
  const float* bc1    = (const float*)d_in[9];
  const float* Wc2    = (const float*)d_in[10];
  const float* bc2    = (const float*)d_in[11];
  float* out = (float*)d_out;

  // workspace layout (u16 units), ~62.3 MB total
  u16* xb    = (u16*)d_ws;                                  // B*S*E
  u16* wpb   = xb    + (size_t)B_ * S_ * E_;                // P*E
  u16* wcb   = wpb   + (size_t)P_ * E_;                     // NK*KW*P*P  [k][t][po][pi]
  u16* wab   = wcb   + (size_t)NK_ * KW_ * P_ * P_;         // S*DL
  u16* projt = wab   + (size_t)S_ * DL_;                    // B*S*P      [b][s][p]
  u16* convb = projt + (size_t)B_ * S_ * P_;                // B*H*S*DL   [b][h][s][l]
  u16* convt = convb + (size_t)B_ * H_ * S_ * DL_;          // B*H*DL*S   [b][h][l][t]
  float* rep_acc = (float*)(convt + (size_t)B_ * H_ * S_ * DL_);

  // conversion passes
  {
    int n4 = (B_ * S_ * E_) / 4;
    cvt4_kernel<<<(n4 + 255) / 256, 256, 0, stream>>>(x, xb, n4);
  }
  {
    int n4 = (P_ * E_) / 4;
    cvt4_kernel<<<(n4 + 255) / 256, 256, 0, stream>>>(Wp, wpb, n4);
  }
  {
    int n4 = (S_ * DL_) / 4;
    cvt4_kernel<<<(n4 + 255) / 256, 256, 0, stream>>>(Wa, wab, n4);
  }
  {
    int n = NK_ * KW_ * P_ * P_;
    cvt_wconv_kernel<<<(n + 255) / 256, 256, 0, stream>>>(Wconv, wcb);
  }
  zero_rep_kernel<<<1, B_ * DL_, 0, stream>>>(rep_acc);

  // 32x32 wave-tiles: B * (P/32) * (S/32) = 4096 waves, 8 waves / block
  const int tiles  = B_ * (P_ / 32) * (S_ / 32);
  const int blocks = tiles / 8;
  proj_kernel<<<blocks, 256, 0, stream>>>(xb, seqlen, wpb, bp, projt);
  conv_kernel<<<blocks, 256, 0, stream>>>(projt, wcb, bconv, convb, convt);
  attn_kernel<<<B_ * H_ * (S_ / 16), 32, 16 * S_ * sizeof(float), stream>>>(
      convb, convt, wab, ba, seqlen, rep_acc);
  head_kernel<<<1, 32, 0, stream>>>(rep_acc, Wc1, bc1, Wc2, bc2, out);
}
